// TokenFuser_57269093925020
// MI455X (gfx1250) — matrix operand order repaired and checked by reading, not compiled, not run on hardware
//
#include <hip/hip_runtime.h>
#include <math.h>

// ---------------------------------------------------------------------------
// MI455X (gfx1250) implementation. Wave32. WMMA bf16 16x16x32 for GEMMs.
// Workspace layout (floats):
//   pos_sum [100352] | pos_sq [100352] | pcsum [8*28*768] | pcmax [8*28*768]
//   m1 [8*8*12544]   | gate [8*8*12544]| tws [8*8*768]    | bias [8*768]
// ---------------------------------------------------------------------------

typedef __attribute__((ext_vector_type(16))) __bf16 v16bf;
typedef __attribute__((ext_vector_type(8)))  float  v8f;
typedef __attribute__((ext_vector_type(4)))  unsigned int v4u;
typedef __attribute__((ext_vector_type(8)))  int v8i_;
typedef __attribute__((ext_vector_type(4)))  int v4i_;

union BfFrag { v16bf v; unsigned int u[8]; uint4 q[2]; };

__device__ __forceinline__ unsigned short f2bf(float f) {
    unsigned int u = __float_as_uint(f);
    u += 0x7fffu + ((u >> 16) & 1u);          // round-to-nearest-even
    return (unsigned short)(u >> 16);
}
__device__ __forceinline__ unsigned int pack2bf(float lo, float hi) {
    return (unsigned int)f2bf(lo) | ((unsigned int)f2bf(hi) << 16);
}

#define NPOS 12544          // 28*28*16 positions per batch
#define DM   768
#define NT   8

#if __has_builtin(__builtin_amdgcn_tensor_load_to_lds) && __has_builtin(__builtin_amdgcn_s_wait_tensorcnt)
#define HAVE_TDM 1
#endif

// ---------------------------------------------------------------------------
// Kernel P: one pass over `original`; per-(b,c) sum & max partials (per h-slice)
// and per-position sum / sum-of-squares (for LayerNorm) via atomics.
// grid (28 h, 3 cgroup, 8 b) x 256 threads (1 thread = 1 channel).
// ---------------------------------------------------------------------------
__global__ void __launch_bounds__(256)
pool_kernel(const float* __restrict__ orig,
            float* __restrict__ pos_sum, float* __restrict__ pos_sq,
            float* __restrict__ pcsum,   float* __restrict__ pcmax)
{
    const int h   = blockIdx.x;
    const int cg  = blockIdx.y;
    const int b   = blockIdx.z;
    const int tid = threadIdx.x;
    const int c   = cg * 256 + tid;
    const int lane = tid & 31;

    __shared__ float s_psum[448];
    __shared__ float s_psq[448];
    for (int i = tid; i < 448; i += 256) { s_psum[i] = 0.f; s_psq[i] = 0.f; }
    __syncthreads();

    const size_t base = ((size_t)(b * 28 + h)) * 28 * 16 * DM;
    float csum = 0.f, cmax = -3.4e38f;
    for (int p = 0; p < 448; ++p) {
        if ((p & 3) == 0)
            __builtin_prefetch(orig + base + (size_t)(p + 8) * DM + c, 0, 0);
        float v = orig[base + (size_t)p * DM + c];
        csum += v;
        cmax  = fmaxf(cmax, v);
        float s = v, q = v * v;
        #pragma unroll
        for (int off = 16; off > 0; off >>= 1) {
            s += __shfl_xor(s, off, 32);
            q += __shfl_xor(q, off, 32);
        }
        if (lane == 0) { atomicAdd(&s_psum[p], s); atomicAdd(&s_psq[p], q); }
    }
    __syncthreads();

    const size_t posbase = ((size_t)(b * 28 + h)) * 448;
    for (int p = tid; p < 448; p += 256) {
        atomicAdd(&pos_sum[posbase + p], s_psum[p]);
        atomicAdd(&pos_sq [posbase + p], s_psq[p]);
    }
    pcsum[((size_t)(b * 28 + h)) * DM + c] = csum;
    pcmax[((size_t)(b * 28 + h)) * DM + c] = cmax;
}

// ---------------------------------------------------------------------------
// Kernel Q: per-batch prep. Token branch (LN -> mix -> LN), ChannelGate MLP,
// WMMA GEMM tw[n,o] = sum_c t[n,c]*conv3_w[o,c]  (M=16 rows, rows 8..15 zero),
// and rank-1 chout[o] = sum_c ch[c]*conv3_w[o,768+c] as plain dots.
// D rows 0..7 -> tws (scaled by BN3 s3); bias[b,o] = s3*(chout-m3)+b3.
// grid 8 x 256 threads (8 waves).
// ---------------------------------------------------------------------------
__global__ void __launch_bounds__(256)
prep_kernel(const float* __restrict__ inputs,
            const float* __restrict__ norm_g, const float* __restrict__ norm_b,
            const float* __restrict__ lin_w,
            const float* __restrict__ pcsum,  const float* __restrict__ pcmax,
            const float* __restrict__ mlp_w1, const float* __restrict__ mlp_b1,
            const float* __restrict__ mlp_w2, const float* __restrict__ mlp_b2,
            const float* __restrict__ conv3_w,
            const float* __restrict__ bn3_g, const float* __restrict__ bn3_b,
            const float* __restrict__ bn3_m, const float* __restrict__ bn3_v,
            float* __restrict__ tws, float* __restrict__ bias)
{
    const int b    = blockIdx.x;
    const int tid  = threadIdx.x;
    const int lane = tid & 31;
    const int wv   = tid >> 5;

    __shared__ float s_avg[DM];
    __shared__ float s_mx[DM];
    __shared__ float s_t0[NT][DM];
    __shared__ __align__(16) unsigned short s_A[16][DM];  // bf16 A tile, rows 8..15 = 0
    __shared__ float s_ch[DM];
    __shared__ float s_h[16];
    __shared__ float s_att[NT];

    // zero A rows 8..15 (rows 0..7 fully overwritten below)
    for (int i = tid; i < 8 * DM / 2; i += 256)
        *(unsigned int*)&s_A[8 + i / 384][(i % 384) * 2] = 0u;

    // 1) finalize spatial avg / max pooling
    for (int c = tid; c < DM; c += 256) {
        float s = 0.f, m = -3.4e38f;
        for (int h = 0; h < 28; ++h) {
            s += pcsum[((size_t)(b * 28 + h)) * DM + c];
            m  = fmaxf(m, pcmax[((size_t)(b * 28 + h)) * DM + c]);
        }
        s_avg[c] = s * (1.f / 12544.f);
        s_mx[c]  = m;
    }

    // 2) LN(inputs[b]); wave wv owns token wv
    {
        const float* x = inputs + ((size_t)b * NT + wv) * DM;
        float s = 0.f, q = 0.f;
        for (int i = 0; i < 24; ++i) { float v = x[lane + 32 * i]; s += v; q += v * v; }
        #pragma unroll
        for (int off = 16; off > 0; off >>= 1) { s += __shfl_xor(s, off, 32); q += __shfl_xor(q, off, 32); }
        float mean = s * (1.f / DM), var = q * (1.f / DM) - mean * mean;
        float rstd = rsqrtf(var + 1e-5f);
        for (int i = 0; i < 24; ++i) {
            int c = lane + 32 * i;
            s_t0[wv][c] = (x[c] - mean) * rstd * norm_g[c] + norm_b[c];
        }
    }
    __syncthreads();

    // 3) token mix (lin_w[j,i]) + second LN -> bf16 A rows 0..7
    {
        float lw[NT];
        #pragma unroll
        for (int i = 0; i < NT; ++i) lw[i] = lin_w[wv * NT + i];
        float tm[24];
        float s = 0.f, q = 0.f;
        for (int i = 0; i < 24; ++i) {
            int c = lane + 32 * i;
            float acc = 0.f;
            #pragma unroll
            for (int k = 0; k < NT; ++k) acc += lw[k] * s_t0[k][c];
            tm[i] = acc; s += acc; q += acc * acc;
        }
        #pragma unroll
        for (int off = 16; off > 0; off >>= 1) { s += __shfl_xor(s, off, 32); q += __shfl_xor(q, off, 32); }
        float mean = s * (1.f / DM), var = q * (1.f / DM) - mean * mean;
        float rstd = rsqrtf(var + 1e-5f);
        for (int i = 0; i < 24; ++i) {
            int c = lane + 32 * i;
            s_A[wv][c] = f2bf((tm[i] - mean) * rstd * norm_g[c] + norm_b[c]);
        }
    }
    __syncthreads();

    // 4) ChannelGate MLP hidden = gelu(pool @ w1^T + b1); att = sigmoid(sum)
    {
        const float* w1 = mlp_w1 + (size_t)wv * DM;
        float sa = 0.f, sm = 0.f;
        for (int i = 0; i < 24; ++i) {
            int c = lane + 32 * i;
            float w = w1[c];
            sa += w * s_avg[c];
            sm += w * s_mx[c];
        }
        #pragma unroll
        for (int off = 16; off > 0; off >>= 1) { sa += __shfl_xor(sa, off, 32); sm += __shfl_xor(sm, off, 32); }
        if (lane == 0) {
            float b1 = mlp_b1[wv];
            float xa = sa + b1, xm = sm + b1;
            s_h[wv]     = 0.5f * xa * (1.f + erff(xa * 0.70710678118f));
            s_h[8 + wv] = 0.5f * xm * (1.f + erff(xm * 0.70710678118f));
        }
    }
    __syncthreads();
    if (tid < NT) {
        float acc = 2.f * mlp_b2[tid];
        #pragma unroll
        for (int k = 0; k < NT; ++k) acc += mlp_w2[tid * NT + k] * (s_h[k] + s_h[8 + k]);
        s_att[tid] = 1.f / (1.f + expf(-acc));
    }
    __syncthreads();

    // 5) ch[c] = sum_n inputs[b,n,c] * att[n]
    for (int c = tid; c < DM; c += 256) {
        float acc = 0.f;
        #pragma unroll
        for (int n = 0; n < NT; ++n) acc += inputs[((size_t)b * NT + n) * DM + c] * s_att[n];
        s_ch[c] = acc;
    }
    __syncthreads();

    // 6) WMMA GEMM: M=16, K=768, N=768 (48 N-tiles, 6 per wave)
    const int hi  = lane >> 4;
    const int col = lane & 15;   // N within tile (and A row for frag loads)
    for (int t = 0; t < 6; ++t) {
        const int nt = wv + 8 * t;
        const int o  = nt * 16 + col;
        v8f acc;
        #pragma unroll
        for (int i = 0; i < 8; ++i) acc[i] = 0.f;

        const float* wp = conv3_w + (size_t)o * 1536;
        for (int k = 0; k < 24; ++k) {
            BfFrag a, bf;
            // A: two contiguous 16B half-fragments (ds_load_b128)
            a.q[0] = *(const uint4*)&s_A[col][k * 32 + hi * 16];
            a.q[1] = *(const uint4*)&s_A[col][k * 32 + 8 + hi * 16];
            // B: conv3_w[o][k*32 + hi*16 .. +15] (global_load_b128 x4 + pack)
            const float* wk = wp + k * 32 + hi * 16;
            float4 w0 = *(const float4*)(wk + 0);
            float4 w1 = *(const float4*)(wk + 4);
            float4 w2 = *(const float4*)(wk + 8);
            float4 w3 = *(const float4*)(wk + 12);
            bf.u[0] = pack2bf(w0.x, w0.y); bf.u[1] = pack2bf(w0.z, w0.w);
            bf.u[2] = pack2bf(w1.x, w1.y); bf.u[3] = pack2bf(w1.z, w1.w);
            bf.u[4] = pack2bf(w2.x, w2.y); bf.u[5] = pack2bf(w2.z, w2.w);
            bf.u[6] = pack2bf(w3.x, w3.y); bf.u[7] = pack2bf(w3.z, w3.w);
            acc = __builtin_amdgcn_wmma_f32_16x16x32_bf16(
                      false, a.v, false, bf.v, (short)0, acc, false, false);
        }

        if (hi == 0) {
            const float s3 = bn3_g[o] * rsqrtf(bn3_v[o] + 1e-5f);
            #pragma unroll
            for (int r = 0; r < 8; ++r)
                tws[((size_t)(b * NT + r)) * DM + o] = s3 * acc[r];
        }
    }
    // 7) rank-1 chout -> bias (plain dots; conv3_w second half)
    for (int o = tid; o < DM; o += 256) {
        const float* wp = conv3_w + (size_t)o * 1536 + DM;
        float acc = 0.f;
        for (int c = 0; c < DM; ++c) acc += s_ch[c] * wp[c];
        float s3 = bn3_g[o] * rsqrtf(bn3_v[o] + 1e-5f);
        bias[(size_t)b * DM + o] = s3 * (acc - bn3_m[o]) + bn3_b[o];
    }
}

// ---------------------------------------------------------------------------
// Kernel G: LN(original) (stats precomputed) + conv1 (768->8) via WMMA bf16,
// fused BN1 + ReLU. One wave per 16-position tile. grid 6272 x 32.
// ---------------------------------------------------------------------------
__global__ void __launch_bounds__(32)
conv1_wmma_kernel(const float* __restrict__ orig,
                  const float* __restrict__ pos_sum, const float* __restrict__ pos_sq,
                  const float* __restrict__ norm_g,  const float* __restrict__ norm_b,
                  const float* __restrict__ conv1_w,
                  const float* __restrict__ bn1_g, const float* __restrict__ bn1_b,
                  const float* __restrict__ bn1_m, const float* __restrict__ bn1_v,
                  float* __restrict__ m1)
{
    __shared__ __align__(16) unsigned short As[16][DM];  // bf16 LN'd acts [pos][c]
    __shared__ __align__(16) unsigned short Bs[16][DM];  // bf16 conv1_w^T [col][c]
    __shared__ float s_mean[16], s_rstd[16];

    const int lane  = threadIdx.x;
    const int tile  = blockIdx.x;           // 0..6271
    const int b     = tile / 784;
    const int ptile = tile % 784;
    const size_t pos0 = (size_t)b * NPOS + (size_t)ptile * 16;

    // stage B (conv1_w is tiny & L2 resident); cols 8..15 zero
    for (int i = lane; i < 8 * DM / 2; i += 32) {
        int c2 = i / 384, cp = (i % 384) * 2;
        *(unsigned int*)&Bs[c2][cp] = pack2bf(conv1_w[c2 * DM + cp], conv1_w[c2 * DM + cp + 1]);
    }
    for (int i = lane; i < 8 * DM / 2; i += 32) {
        int c2 = 8 + i / 384, cp = (i % 384) * 2;
        *(unsigned int*)&Bs[c2][cp] = 0u;
    }
    if (lane < 16) {
        float s = pos_sum[pos0 + lane];
        float q = pos_sq [pos0 + lane];
        float mean = s * (1.f / DM);
        float var  = q * (1.f / DM) - mean * mean;
        s_mean[lane] = mean;
        s_rstd[lane] = rsqrtf(var + 1e-5f);
    }
    __syncthreads();

    // stream 16x768 fp32 -> LN -> bf16 into LDS (b128 coalesced loads)
    const float* src = orig + pos0 * DM;
    for (int i = 0; i < 96; ++i) {
        int flat = i * 128 + lane * 4;      // = p*768 + c
        if ((i & 3) == 0)
            __builtin_prefetch(src + flat + 2048, 0, 0);
        int p = flat / DM;
        int c = flat % DM;
        float4 v  = *(const float4*)(src + flat);
        float4 g  = *(const float4*)(norm_g + c);
        float4 bb = *(const float4*)(norm_b + c);
        float mean = s_mean[p], rstd = s_rstd[p];
        uint2 pk;
        pk.x = pack2bf((v.x - mean) * rstd * g.x + bb.x,
                       (v.y - mean) * rstd * g.y + bb.y);
        pk.y = pack2bf((v.z - mean) * rstd * g.z + bb.z,
                       (v.w - mean) * rstd * g.w + bb.w);
        *(uint2*)&As[p][c] = pk;
    }
    __syncthreads();

    // WMMA: 24 K-steps of 16x16x32 bf16; b128 LDS fragment loads
    const int hi  = lane >> 4;
    const int col = lane & 15;
    v8f acc;
    #pragma unroll
    for (int i = 0; i < 8; ++i) acc[i] = 0.f;
    for (int k = 0; k < 24; ++k) {
        BfFrag a, bf;
        a.q[0]  = *(const uint4*)&As[col][k * 32 + hi * 8];
        a.q[1]  = *(const uint4*)&As[col][k * 32 + 16 + hi * 8];
        bf.q[0] = *(const uint4*)&Bs[col][k * 32 + hi * 8];
        bf.q[1] = *(const uint4*)&Bs[col][k * 32 + 16 + hi * 8];
        acc = __builtin_amdgcn_wmma_f32_16x16x32_bf16(
                  false, a.v, false, bf.v, (short)0, acc, false, false);
    }

    // BN1 + ReLU, write m1[b][n][pos]
    if (col < 8) {
        float s1 = bn1_g[col] * rsqrtf(bn1_v[col] + 1e-5f);
        float mn = bn1_m[col], bb1 = bn1_b[col];
        #pragma unroll
        for (int r = 0; r < 8; ++r) {
            int pm = r + 8 * hi;
            float val = fmaxf((acc[r] - mn) * s1 + bb1, 0.f);
            m1[((size_t)(b * NT + col)) * NPOS + ptile * 16 + pm] = val;
        }
    }
}

// ---------------------------------------------------------------------------
// Kernel C2: 3x3x3 conv (8->8 ch, SAME) + BN2 + ReLU + sigmoid -> gate.
// grid (28 h, 8 n_out, 8 b) x 256 threads.
// ---------------------------------------------------------------------------
__global__ void __launch_bounds__(256)
conv2_kernel(const float* __restrict__ m1, const float* __restrict__ cw,
             const float* __restrict__ bn2_g, const float* __restrict__ bn2_b,
             const float* __restrict__ bn2_m, const float* __restrict__ bn2_v,
             float* __restrict__ gate)
{
    const int h  = blockIdx.x;
    const int no = blockIdx.y;
    const int b  = blockIdx.z;
    const int tid = threadIdx.x;

    __shared__ float s_w[NT * 27];
    if (tid < NT * 27) s_w[tid] = cw[no * NT * 27 + tid];
    __syncthreads();

    const float s2 = bn2_g[no] * rsqrtf(bn2_v[no] + 1e-5f);
    const float mn = bn2_m[no], bb = bn2_b[no];

    for (int idx = tid; idx < 448; idx += 256) {
        int w  = idx >> 4;
        int dd = idx & 15;
        float acc = 0.f;
        for (int ni = 0; ni < NT; ++ni) {
            const float* src = m1 + ((size_t)(b * NT + ni)) * NPOS;
            #pragma unroll
            for (int a = 0; a < 3; ++a) {
                int hh = h + a - 1;
                if (hh < 0 || hh >= 28) continue;
                #pragma unroll
                for (int e = 0; e < 3; ++e) {
                    int ww = w + e - 1;
                    if (ww < 0 || ww >= 28) continue;
                    #pragma unroll
                    for (int f = 0; f < 3; ++f) {
                        int d2 = dd + f - 1;
                        if (d2 < 0 || d2 >= 16) continue;
                        acc += src[(hh * 28 + ww) * 16 + d2] * s_w[((ni * 3 + a) * 3 + e) * 3 + f];
                    }
                }
            }
        }
        float v = fmaxf((acc - mn) * s2 + bb, 0.f);
        gate[((size_t)(b * NT + no)) * NPOS + h * 448 + idx] = 1.f / (1.f + expf(-v));
    }
}

// ---------------------------------------------------------------------------
// Kernel F: out[pos,o] = relu( sum_n gate[n,pos]*tws[n,o] + bias[o] )
// grid (784 ptile, 8 b) x 256 threads; float4 coalesced stores (308 MB).
// tws tile (24KB) staged to LDS via the Tensor Data Mover when available.
// ---------------------------------------------------------------------------
__global__ void __launch_bounds__(256)
fuse_kernel(const float* __restrict__ tws, const float* __restrict__ bias,
            const float* __restrict__ gate, float* __restrict__ out)
{
    const int pt  = blockIdx.x;
    const int b   = blockIdx.y;
    const int tid = threadIdx.x;

    __shared__ __align__(16) float s_tws[NT][DM];
    __shared__ float s_bias[DM];
    __shared__ float s_g[NT][16];

#ifdef HAVE_TDM
    if (tid < 32) {
        // TDM: contiguous 1D tile, 6144 elements of 4B (24 KB) -> LDS
        unsigned lds = (unsigned)(size_t)&s_tws[0][0];
        unsigned long long ga = (unsigned long long)(const void*)(tws + (size_t)b * NT * DM);
        v4u g0 = { 1u,                                    // count=1, user mode
                   lds,                                   // lds_addr (bytes)
                   (unsigned)ga,                          // global_addr[31:0]
                   (unsigned)((ga >> 32) & 0x1FFFFFFu) | 0x80000000u }; // addr[56:32] | type=2
        v8i_ g1 = { (int)0x00020000u,                     // data_size=2 (4B)
                    (int)(6144u << 16),                   // tensor_dim0 lo16 in [31:16]
                    (int)(1u << 16),                      // tensor_dim0 hi=0 | tensor_dim1=1
                    (int)(6144u << 16),                   // tensor_dim1 hi=0 | tile_dim0=6144
                    1,                                    // tile_dim1=1, tile_dim2=0
                    6144,                                 // tensor_dim0_stride lo32
                    0, 0 };
        v4i_ z4 = { 0, 0, 0, 0 };
        v8i_ z8 = { 0, 0, 0, 0, 0, 0, 0, 0 };
        __builtin_amdgcn_tensor_load_to_lds(g0, g1, z4, z4, z8, 0);
        __builtin_amdgcn_s_wait_tensorcnt(0);
    }
#else
    for (int i = tid; i < NT * DM; i += 256)
        s_tws[i / DM][i % DM] = tws[(size_t)b * NT * DM + i];
#endif
    for (int i = tid; i < DM; i += 256)
        s_bias[i] = bias[(size_t)b * DM + i];
    if (tid < 128) {
        int n = tid >> 4, p = tid & 15;
        s_g[n][p] = gate[((size_t)(b * NT + n)) * NPOS + pt * 16 + p];
    }
    __syncthreads();

    float* dst = out + ((size_t)b * NPOS + (size_t)pt * 16) * DM;
    #pragma unroll
    for (int it = 0; it < 12; ++it) {
        int idx = it * 256 + tid;          // 0..3071
        int p   = idx / 192;               // position in tile
        int o   = (idx % 192) * 4;
        float4 acc = *(const float4*)&s_bias[o];
        #pragma unroll
        for (int n = 0; n < NT; ++n) {
            float g = s_g[n][p];
            float4 t4 = *(const float4*)&s_tws[n][o];
            acc.x += g * t4.x; acc.y += g * t4.y;
            acc.z += g * t4.z; acc.w += g * t4.w;
        }
        acc.x = fmaxf(acc.x, 0.f); acc.y = fmaxf(acc.y, 0.f);
        acc.z = fmaxf(acc.z, 0.f); acc.w = fmaxf(acc.w, 0.f);
        *(float4*)(dst + (size_t)p * DM + o) = acc;
    }
}

// ---------------------------------------------------------------------------
extern "C" void kernel_launch(void* const* d_in, const int* in_sizes, int n_in,
                              void* d_out, int out_size, void* d_ws, size_t ws_size,
                              hipStream_t stream)
{
    const float* inputs   = (const float*)d_in[0];
    const float* original = (const float*)d_in[1];
    const float* norm_g   = (const float*)d_in[2];
    const float* norm_b   = (const float*)d_in[3];
    const float* lin_w    = (const float*)d_in[4];
    const float* conv1_w  = (const float*)d_in[5];
    const float* bn1_g    = (const float*)d_in[6];
    const float* bn1_b    = (const float*)d_in[7];
    const float* bn1_m    = (const float*)d_in[8];
    const float* bn1_v    = (const float*)d_in[9];
    const float* conv2_w  = (const float*)d_in[10];
    const float* bn2_g    = (const float*)d_in[11];
    const float* bn2_b    = (const float*)d_in[12];
    const float* bn2_m    = (const float*)d_in[13];
    const float* bn2_v    = (const float*)d_in[14];
    const float* conv3_w  = (const float*)d_in[15];
    const float* bn3_g    = (const float*)d_in[16];
    const float* bn3_b    = (const float*)d_in[17];
    const float* bn3_m    = (const float*)d_in[18];
    const float* bn3_v    = (const float*)d_in[19];
    const float* mlp_w1   = (const float*)d_in[20];
    const float* mlp_b1   = (const float*)d_in[21];
    const float* mlp_w2   = (const float*)d_in[22];
    const float* mlp_b2   = (const float*)d_in[23];
    // d_in[24] = reduce_w : unused (reference uses it only for shape)

    float* ws      = (float*)d_ws;
    float* pos_sum = ws;                        // 100352
    float* pos_sq  = ws + 100352;               // 100352
    float* pcsum   = ws + 200704;               // 172032
    float* pcmax   = ws + 372736;               // 172032
    float* m1      = ws + 544768;               // 802816
    float* gate    = ws + 1347584;              // 802816
    float* tws     = ws + 2150400;              // 49152
    float* biasv   = ws + 2199552;              // 6144

    (void)hipMemsetAsync(pos_sum, 0, 2 * 100352 * sizeof(float), stream);

    pool_kernel<<<dim3(28, 3, 8), 256, 0, stream>>>(original, pos_sum, pos_sq, pcsum, pcmax);

    prep_kernel<<<8, 256, 0, stream>>>(inputs, norm_g, norm_b, lin_w, pcsum, pcmax,
                                       mlp_w1, mlp_b1, mlp_w2, mlp_b2,
                                       conv3_w, bn3_g, bn3_b, bn3_m, bn3_v,
                                       tws, biasv);

    conv1_wmma_kernel<<<6272, 32, 0, stream>>>(original, pos_sum, pos_sq, norm_g, norm_b,
                                               conv1_w, bn1_g, bn1_b, bn1_m, bn1_v, m1);

    conv2_kernel<<<dim3(28, 8, 8), 256, 0, stream>>>(m1, conv2_w, bn2_g, bn2_b, bn2_m, bn2_v, gate);

    fuse_kernel<<<dim3(784, 8), 256, 0, stream>>>(tws, biasv, gate, (float*)d_out);
}
